// neural_symbolc_base_22754736734541
// MI455X (gfx1250) — matrix operand order repaired and checked by reading, not compiled, not run on hardware
//
#include <hip/hip_runtime.h>
#include <hip/hip_bf16.h>

#define BATCH 64
#define CH    256
#define LEN   4096
#define MROWS 112      // 64 multi rows + 48 down rows, fused GEMM
#define NFUN  48
#define EPSV  1e-5f
#define NTI   8        // 16-wide n-tiles per block -> 32 blocks per batch
#define LDSTR 130      // LDS row stride in dwords (even => conflict-free pairs)

typedef __attribute__((ext_vector_type(8)))  float        v8f;
typedef __attribute__((ext_vector_type(8)))  unsigned int v8u;
typedef __attribute__((ext_vector_type(16))) __bf16       v16bf;

// Pack two floats into two RNE bf16s in one dword (low half = f0).
static __device__ __forceinline__ unsigned pack_bf16(float f0, float f1) {
#if __has_builtin(__builtin_amdgcn_cvt_pk_bf16_f32)
  typedef __attribute__((ext_vector_type(2))) __bf16 v2bf;
  v2bf p = __builtin_amdgcn_cvt_pk_bf16_f32(f0, f1);
  return __builtin_bit_cast(unsigned, p);
#else
  unsigned u0 = __float_as_uint(f0), u1 = __float_as_uint(f1);
  unsigned r0 = u0 + 0x7FFFu + ((u0 >> 16) & 1u);
  unsigned r1 = u1 + 0x7FFFu + ((u1 >> 16) & 1u);
  return (r0 >> 16) | (r1 & 0xFFFF0000u);
#endif
}
static __device__ __forceinline__ unsigned bf16_rne(float f) {
  unsigned u = __float_as_uint(f);
  return (u + 0x7FFFu + ((u >> 16) & 1u)) >> 16;
}
static __device__ __forceinline__ float bf16_up(unsigned h) {
  return __uint_as_float(h << 16);
}

// ---------------------------------------------------------------------------
// Kernel 1: per-(b,c) mean / rstd of x over L. Also zeroes the 6144-float
// sum/sumsq accumulator region (blocks 0..23).
// ---------------------------------------------------------------------------
__global__ __launch_bounds__(256) void k_xstats(const float* __restrict__ x,
                                                float* __restrict__ meanv,
                                                float* __restrict__ rstdv,
                                                float* __restrict__ sums) {
  __shared__ float rs[256], rq[256];
  int blk = blockIdx.x;            // b*CH + c
  int t = threadIdx.x;
  if (blk < 24) sums[blk * 256 + t] = 0.f;
  const float4* xp = (const float4*)(x + (size_t)blk * LEN);
  float s = 0.f, q = 0.f;
#pragma unroll
  for (int k = 0; k < 4; ++k) {
    float4 f = xp[t + 256 * k];
    s += f.x + f.y + f.z + f.w;
    q += f.x * f.x + f.y * f.y + f.z * f.z + f.w * f.w;
  }
  rs[t] = s; rq[t] = q;
  __syncthreads();
  for (int off = 128; off > 0; off >>= 1) {
    if (t < off) { rs[t] += rs[t + off]; rq[t] += rq[t + off]; }
    __syncthreads();
  }
  if (t == 0) {
    float m = rs[0] * (1.f / LEN);
    float v = rq[0] * (1.f / LEN) - m * m;
    meanv[blk] = m;
    rstdv[blk] = rsqrtf(fmaxf(v, 0.f) + EPSV);
  }
}

// ---------------------------------------------------------------------------
// Kernel 2: softmax of w_channel over the output-channel dim (64 rows).
// ---------------------------------------------------------------------------
__global__ __launch_bounds__(256) void k_softmax(const float* __restrict__ wch,
                                                 float* __restrict__ weff) {
  int c = threadIdx.x;
  float mx = -1e30f;
  for (int o = 0; o < 64; ++o) mx = fmaxf(mx, wch[o * CH + c]);
  float e[64];
  float sum = 0.f;
  for (int o = 0; o < 64; ++o) { e[o] = expf(wch[o * CH + c] - mx); sum += e[o]; }
  float inv = 1.f / sum;
  for (int o = 0; o < 64; ++o) weff[o * CH + c] = e[o] * inv;
}

// ---------------------------------------------------------------------------
// Kernel 3: per-batch effective weights. Row o<64 -> w_eff, o>=64 -> w_down.
// W'[b,o,c] = w[o,c]*rstd[b,c], split into bf16 hi/lo; bias = -sum W'*mean.
// ---------------------------------------------------------------------------
__global__ __launch_bounds__(64) void k_wprep(const float* __restrict__ weff,
                                              const float* __restrict__ wdown,
                                              const float* __restrict__ meanv,
                                              const float* __restrict__ rstdv,
                                              unsigned short* __restrict__ whi,
                                              unsigned short* __restrict__ wlo,
                                              float* __restrict__ biasv) {
  __shared__ float red[64];
  int blk = blockIdx.x;                 // b*MROWS + o
  int b = blk / MROWS, o = blk % MROWS;
  const float* src = (o < 64) ? (weff + (size_t)o * CH)
                              : (wdown + (size_t)(o - 64) * CH);
  int t = threadIdx.x;
  size_t rowoff = (size_t)blk * CH;
  float bacc = 0.f;
#pragma unroll
  for (int i = 0; i < 4; ++i) {
    int c = t + 64 * i;
    float W = src[c] * rstdv[b * CH + c];
    unsigned h = bf16_rne(W);
    whi[rowoff + c] = (unsigned short)h;
    wlo[rowoff + c] = (unsigned short)bf16_rne(W - bf16_up(h));
    bacc += W * meanv[b * CH + c];
  }
  red[t] = bacc;
  __syncthreads();
  for (int off = 32; off > 0; off >>= 1) {
    if (t < off) red[t] += red[t + off];
    __syncthreads();
  }
  if (t == 0) biasv[blk] = -red[0];
}

// ---------------------------------------------------------------------------
// Kernel 4: fused 112x256x4096 per-batch GEMM on bf16 WMMA (hi/lo split,
// 3 WMMAs per k-tile) + symbolic function application + partial stats.
// x is converted fp32 -> packed bf16 hi/lo ONCE during LDS staging; the
// k-loop is then pure ds_load + v_wmma.
// ---------------------------------------------------------------------------
__global__ __launch_bounds__(256) void k_gemm_apply(
    const float* __restrict__ x,
    const unsigned short* __restrict__ whi,
    const unsigned short* __restrict__ wlo,
    const float* __restrict__ biasv,
    float* __restrict__ img,           // [B][48][LEN] raw function outputs
    float* __restrict__ outp,          // [B][48][LEN] skip (down-conv) part
    float* __restrict__ sums)          // [0..3071]=sum, [3072..6143]=sumsq
{
  __shared__ unsigned xh[16][LDSTR];   // packed bf16-hi pairs  [n][cpair]
  __shared__ unsigned xl[16][LDSTR];   // packed bf16-lo pairs
  int blk = blockIdx.x;
  int b = blk >> 5;                    // 32 blocks per batch
  int lblk = blk & 31;
  int l0 = lblk * (16 * NTI);
  int tid = threadIdx.x;
  int wave = tid >> 5;
  int lane = tid & 31;
  int half = lane >> 4;
  int ln = lane & 15;
  int mt = wave;                       // m-tile index for waves 0..6

  // --- preload A fragments (ISA 16-bit A 16x32 layout) and bias -----------
  v8u ahi[8], alo[8];
  float bia[8];
  if (mt < 7) {
    size_t row = ((size_t)b * MROWS + mt * 16 + ln) * CH;
#pragma unroll
    for (int kt = 0; kt < 8; ++kt) {
#pragma unroll
      for (int v = 0; v < 8; ++v) {
        int k = kt * 32 + ((v < 4) ? 2 * v : 2 * v + 8) + 8 * half;
        ahi[kt][v] = (unsigned)whi[row + k] | ((unsigned)whi[row + k + 1] << 16);
        alo[kt][v] = (unsigned)wlo[row + k] | ((unsigned)wlo[row + k + 1] << 16);
      }
    }
#pragma unroll
    for (int r = 0; r < 8; ++r)
      bia[r] = biasv[b * MROWS + mt * 16 + 8 * half + r];
  }

  float s1[6] = {0, 0, 0, 0, 0, 0};
  float s2[6] = {0, 0, 0, 0, 0, 0};

  for (int nt = 0; nt < NTI; ++nt) {
    int lb = l0 + nt * 16;
    __syncthreads();
    {   // stage x[b, :, lb..lb+15] -> packed bf16 hi/lo pairs, all 256 threads
      int n = tid & 15, cp0 = tid >> 4;           // cp0 in 0..15
      const float* xp = x + ((size_t)b * CH) * LEN + lb + n;
#pragma unroll
      for (int i = 0; i < 8; ++i) {
        int cp = cp0 + 16 * i;                    // pair index 0..127
        int c = 2 * cp;
        float f0 = xp[(size_t)c * LEN];
        float f1 = xp[(size_t)(c + 1) * LEN];
        unsigned hi = pack_bf16(f0, f1);
        float h0 = __uint_as_float(hi << 16);
        float h1 = __uint_as_float(hi & 0xFFFF0000u);
        unsigned lo = pack_bf16(f0 - h0, f1 - h1);
        xh[n][cp] = hi;
        xl[n][cp] = lo;
      }
      if (nt + 1 < NTI) __builtin_prefetch(xp + 16, 0, 1);   // next n-tile
    }
    __syncthreads();
    if (mt < 7) {
      v8f acc;
#pragma unroll
      for (int r = 0; r < 8; ++r) acc[r] = bia[r];   // bias as C input
#pragma unroll
      for (int kt = 0; kt < 8; ++kt) {
        int cpb = kt * 16 + 4 * half;            // pair base for this k-tile (table order)
        v8u bhv, blv;
#pragma unroll
        for (int v = 0; v < 8; ++v) {            // pairs cpb+0..3, cpb+8..11
          bhv[v] = xh[ln][cpb + ((v < 4) ? v : v + 4)];
          blv[v] = xl[ln][cpb + ((v < 4) ? v : v + 4)];
        }
        acc = __builtin_amdgcn_wmma_f32_16x16x32_bf16(
            false, __builtin_bit_cast(v16bf, ahi[kt]),
            false, __builtin_bit_cast(v16bf, bhv),
            (short)0, acc, false, false);
        acc = __builtin_amdgcn_wmma_f32_16x16x32_bf16(
            false, __builtin_bit_cast(v16bf, alo[kt]),
            false, __builtin_bit_cast(v16bf, bhv),
            (short)0, acc, false, false);
        acc = __builtin_amdgcn_wmma_f32_16x16x32_bf16(
            false, __builtin_bit_cast(v16bf, ahi[kt]),
            false, __builtin_bit_cast(v16bf, blv),
            (short)0, acc, false, false);
      }

      int l = lb + ln;
      if (mt < 4) {
        // this lane's 8 rows == one function group of 8 multi channels
        float im[6];
        im[0] = acc[0] + acc[1];     // add
        im[1] = acc[2] * acc[3];     // mul
        im[2] = sinf(acc[4]);
        im[3] = cosf(acc[5]);
        im[4] = tanhf(acc[6]);
        im[5] = acc[7];              // identity
        int g = 2 * mt + half;
        float* ip = img + (((size_t)b * NFUN) + 6 * g) * LEN + l;
#pragma unroll
        for (int k = 0; k < 6; ++k) {
          float v = im[k];
          ip[(size_t)k * LEN] = v;
          s1[k] += v;
          s2[k] += v * v;
        }
      } else {
        int o48 = (mt - 4) * 16 + 8 * half;   // down-conv row
        float* op = outp + (((size_t)b * NFUN) + o48) * LEN + l;
#pragma unroll
        for (int r = 0; r < 8; ++r) op[(size_t)r * LEN] = acc[r];
      }
    }
  }

  // reduce partial sums across the 16 lanes of each half, then atomics
  if (mt < 4) {
#pragma unroll
    for (int k = 0; k < 6; ++k) {
      float a = s1[k], q = s2[k];
#pragma unroll
      for (int m = 8; m >= 1; m >>= 1) {
        a += __shfl_xor(a, m, 32);
        q += __shfl_xor(q, m, 32);
      }
      if (ln == 0) {
        int j = 6 * (2 * mt + half) + k;
        atomicAdd(&sums[b * NFUN + j], a);
        atomicAdd(&sums[3072 + b * NFUN + j], q);
      }
    }
  }
}

// ---------------------------------------------------------------------------
// Kernel 5: finalize. One block per (b,j): closed-form cumulative norm scale
// (channel j is instance-normed 48-j times; after the first, mean stays 0 and
// var follows v <- v/(v+eps)), then out = (img-mean)*scale + skip.
// ---------------------------------------------------------------------------
__global__ __launch_bounds__(256) void k_finalize(const float* __restrict__ img,
                                                  const float* __restrict__ sums,
                                                  float* __restrict__ outp) {
  int blk = blockIdx.x;               // b*48 + j
  int j = blk % NFUN;
  float m = sums[blk] * (1.f / LEN);
  float var = fmaxf(sums[3072 + blk] * (1.f / LEN) - m * m, 0.f);
  float scale = rsqrtf(var + EPSV);
  float v = var / (var + EPSV);
  int reps = NFUN - j;
  for (int tt = 1; tt < reps; ++tt) {
    scale *= rsqrtf(v + EPSV);
    v = v / (v + EPSV);
  }
  const float4* ip = (const float4*)(img + (size_t)blk * LEN);
  float4* op = (float4*)(outp + (size_t)blk * LEN);
  int t = threadIdx.x;
#pragma unroll
  for (int k = 0; k < 4; ++k) {
    float4 a = ip[t + 256 * k];
    float4 o = op[t + 256 * k];
    o.x += (a.x - m) * scale;
    o.y += (a.y - m) * scale;
    o.z += (a.z - m) * scale;
    o.w += (a.w - m) * scale;
    op[t + 256 * k] = o;
  }
}

// ---------------------------------------------------------------------------
extern "C" void kernel_launch(void* const* d_in, const int* in_sizes, int n_in,
                              void* d_out, int out_size, void* d_ws, size_t ws_size,
                              hipStream_t stream) {
  (void)in_sizes; (void)n_in; (void)out_size; (void)ws_size;
  const float* x   = (const float*)d_in[0];
  const float* wch = (const float*)d_in[1];
  const float* wdn = (const float*)d_in[2];
  float* outp = (float*)d_out;

  char* ws = (char*)d_ws;
  size_t off = 0;
  auto take = [&](size_t bytes) -> void* {
    void* p = (void*)(ws + off);
    off += (bytes + 255) & ~(size_t)255;
    return p;
  };
  float* meanv = (float*)take((size_t)BATCH * CH * 4);
  float* rstdv = (float*)take((size_t)BATCH * CH * 4);
  float* weff  = (float*)take((size_t)64 * CH * 4);
  unsigned short* whi = (unsigned short*)take((size_t)BATCH * MROWS * CH * 2);
  unsigned short* wlo = (unsigned short*)take((size_t)BATCH * MROWS * CH * 2);
  float* biasv = (float*)take((size_t)BATCH * MROWS * 4);
  float* sums  = (float*)take((size_t)2 * BATCH * NFUN * 4);
  float* img   = (float*)take((size_t)BATCH * NFUN * LEN * 4);

  k_xstats    <<<BATCH * CH, 256, 0, stream>>>(x, meanv, rstdv, sums);
  k_softmax   <<<1, 256, 0, stream>>>(wch, weff);
  k_wprep     <<<BATCH * MROWS, 64, 0, stream>>>(weff, wdn, meanv, rstdv, whi, wlo, biasv);
  k_gemm_apply<<<BATCH * 32, 256, 0, stream>>>(x, whi, wlo, biasv, img, outp, sums);
  k_finalize  <<<BATCH * NFUN, 256, 0, stream>>>(img, sums, outp);
}
